// GAT_Layer_678604833488
// MI455X (gfx1250) — matrix operand order repaired and checked by reading, not compiled, not run on hardware
//
#include <hip/hip_runtime.h>

typedef __attribute__((ext_vector_type(16))) _Float16 v16h;
typedef __attribute__((ext_vector_type(8)))  float    v8f;
typedef __attribute__((ext_vector_type(2)))  __fp16   fp16x2;

#define NN     8192
#define FIN    256
#define FOUT   64
#define ALPHA  0.2f
#define NEGINF (-9.0e15f)

union ABU { unsigned u[8]; uint2 u2[4]; v16h v; };

// Single-instruction pack: v_cvt_pk_rtz_f16_f32
__device__ __forceinline__ unsigned pack_f16x2(float a, float b) {
    union { fp16x2 h; unsigned u; } z;
    z.h = __builtin_amdgcn_cvt_pkrtz(a, b);
    return z.u;
}

// ---------------- Kernel 1: Wh = h @ W ; fs = Wh@a1 ; fd = Wh@a2 ----------------
__global__ __launch_bounds__(64) void k_wh(const float* __restrict__ h,
                                           const float* __restrict__ W,
                                           const float* __restrict__ a,
                                           float* __restrict__ Wh,
                                           float* __restrict__ fs,
                                           float* __restrict__ fd) {
    const int i = blockIdx.x;      // node row
    const int n = threadIdx.x;     // 0..63 output feature
    const float* hrow = h + i * FIN;
    float acc = 0.f;
#pragma unroll 8
    for (int k = 0; k < FIN; ++k) acc = fmaf(hrow[k], W[k * FOUT + n], acc);
    Wh[i * FOUT + n] = acc;

    __shared__ float2 red[64];
    float2 me; me.x = acc * a[n]; me.y = acc * a[FOUT + n];
    red[n] = me;
    __syncthreads();
    for (int off = 32; off > 0; off >>= 1) {
        if (n < off) { red[n].x += red[n + off].x; red[n].y += red[n + off].y; }
        __syncthreads();
    }
    if (n == 0) { fs[i] = red[0].x; fd[i] = red[0].y; }
}

// -------- Kernel 2: pre-swizzle Wh (f32) into f16 WMMA B-operand layout --------
// Layout: [nt(4)][kb(256)][lane(32)][8 dwords]; lane: n=lane&15, hi=lane>>4,
// dword v holds K = {2v, 2v+1} + 16*hi of block kb; value = Wh[kb*32+K][nt*16+n].
__global__ __launch_bounds__(256) void k_swizzle_b(const float* __restrict__ Wh,
                                                   unsigned* __restrict__ Wb) {
    const int t = blockIdx.x * blockDim.x + threadIdx.x;
    if (t >= 4 * 256 * 32) return;
    const int lane = t & 31;
    const int kb   = (t >> 5) & 255;
    const int nt   = t >> 13;
    const int hi   = lane >> 4;
    const int col  = nt * 16 + (lane & 15);
    const int k0   = kb * 32 + hi * 16;
    unsigned o[8];
#pragma unroll
    for (int v = 0; v < 8; ++v)
        o[v] = pack_f16x2(Wh[(k0 + 2 * v) * FOUT + col],
                          Wh[(k0 + 2 * v + 1) * FOUT + col]);
    uint4* dst = (uint4*)(Wb + (size_t)t * 8);
    uint4 q0; q0.x = o[0]; q0.y = o[1]; q0.z = o[2]; q0.w = o[3];
    uint4 q1; q1.x = o[4]; q1.y = o[5]; q1.z = o[6]; q1.w = o[7];
    dst[0] = q0; dst[1] = q1;
}

// ---------------- Kernel 3: fused masked-softmax + attention @ Wh ----------------
// 16 output rows per block, 256 threads = 8 waves.
__global__ __launch_bounds__(256) void k_attn(const int* __restrict__ adj,
                                              const unsigned* __restrict__ Wb,
                                              const float* __restrict__ fsg,
                                              const float* __restrict__ fdg,
                                              float* __restrict__ out) {
    __shared__ float  s_fs[16];
    __shared__ float  s_m[16];
    __shared__ float  s_inv[16];
    __shared__ float2 s_red[256];
    __shared__ unsigned s_A[2][32 * 10];   // A tiles, 40B/lane padded stride
    __shared__ float  s_acc[8][32][8];     // per-wave partial accumulators

    const int i0 = blockIdx.x * 16;
    const int t  = threadIdx.x;

    if (t < 16) s_fs[t] = fsg[i0 + t];
    __syncthreads();

    // ---------- Pass A: per-row online masked softmax stats ----------
    {
        const int row  = t >> 4;           // 0..15
        const int slot = t & 15;           // 0..15, 4 consecutive j each
        const float fsv = s_fs[row];
        const int* arow = adj + (size_t)(i0 + row) * NN;
        float m = -3.0e38f, s = 0.f;
        for (int jb = 0; jb < NN; jb += 64) {
            const int j = jb + slot * 4;
            const int4   av = *(const int4*)(arow + j);
            const float4 fv = *(const float4*)(fdg + j);
            float e0 = fsv + fv.x; e0 = e0 > 0.f ? e0 : ALPHA * e0; if (av.x <= 0) e0 = NEGINF;
            float e1 = fsv + fv.y; e1 = e1 > 0.f ? e1 : ALPHA * e1; if (av.y <= 0) e1 = NEGINF;
            float e2 = fsv + fv.z; e2 = e2 > 0.f ? e2 : ALPHA * e2; if (av.z <= 0) e2 = NEGINF;
            float e3 = fsv + fv.w; e3 = e3 > 0.f ? e3 : ALPHA * e3; if (av.w <= 0) e3 = NEGINF;
            const float m4 = fmaxf(fmaxf(e0, e1), fmaxf(e2, e3));
            const float s4 = __expf(e0 - m4) + __expf(e1 - m4) +
                             __expf(e2 - m4) + __expf(e3 - m4);
            const float nm = fmaxf(m, m4);
            s = s * __expf(m - nm) + s4 * __expf(m4 - nm);
            m = nm;
        }
        float2 p; p.x = m; p.y = s;
        s_red[t] = p;
    }
    __syncthreads();
    if (t < 16) {
        float m = s_red[t * 16].x, s = s_red[t * 16].y;
        for (int q = 1; q < 16; ++q) {
            const float2 p = s_red[t * 16 + q];
            const float nm = fmaxf(m, p.x);
            s = s * __expf(m - nm) + p.y * __expf(p.x - nm);
            m = nm;
        }
        s_m[t] = m;
        s_inv[t] = 1.0f / s;
    }
    __syncthreads();

    // ---------- Pass B: P(f16) staged to LDS in A-layout, WMMA accumulate ----------
    const int lane  = t & 31;
    const int g     = (t >> 5) & 3;       // staging group within tile
    const int khalf = t >> 7;             // staging K-half
    const int wv    = t >> 5;             // wave 0..7
    const int nt    = wv & 3;             // wave's N tile (features nt*16..)
    const int wkh   = wv >> 2;            // wave's K-half
    const int M     = lane & 15;
    const int hi    = lane >> 4;
    const int Kbase = g * 4 + hi * 16;

    const int* arowM = adj + (size_t)(i0 + M) * NN;
    const float fsM = s_fs[M];
    const float mM  = s_m[M];
    unsigned* aslot = &s_A[khalf][lane * 10 + g * 2];
    const unsigned* ard = &s_A[wkh][lane * 10];
    const uint4* Wbq = (const uint4*)Wb;
    size_t boff = ((size_t)(nt * 256 + wkh) * 32 + lane) * 2;

    v8f acc = {};
    for (int it = 0; it < 128; ++it) {
        // stage 16x32 P sub-tile (this thread: 4 consecutive K)
        const int j = it * 64 + khalf * 32 + Kbase;
        const int4   av = *(const int4*)(arowM + j);
        const float4 fv = *(const float4*)(fdg + j);
        float e0 = fsM + fv.x; e0 = e0 > 0.f ? e0 : ALPHA * e0; if (av.x <= 0) e0 = NEGINF;
        float e1 = fsM + fv.y; e1 = e1 > 0.f ? e1 : ALPHA * e1; if (av.y <= 0) e1 = NEGINF;
        float e2 = fsM + fv.z; e2 = e2 > 0.f ? e2 : ALPHA * e2; if (av.z <= 0) e2 = NEGINF;
        float e3 = fsM + fv.w; e3 = e3 > 0.f ? e3 : ALPHA * e3; if (av.w <= 0) e3 = NEGINF;
        const float p0 = __expf(e0 - mM), p1 = __expf(e1 - mM);
        const float p2 = __expf(e2 - mM), p3 = __expf(e3 - mM);
        uint2 w; w.x = pack_f16x2(p0, p1); w.y = pack_f16x2(p2, p3);
        *(uint2*)aslot = w;
        __syncthreads();

        // A from LDS (conflict-free padded stride), B from pre-swizzled global
        ABU au;
        au.u2[0] = *(const uint2*)(ard + 0);
        au.u2[1] = *(const uint2*)(ard + 2);
        au.u2[2] = *(const uint2*)(ard + 4);
        au.u2[3] = *(const uint2*)(ard + 6);
        const uint4 b0 = Wbq[boff], b1 = Wbq[boff + 1];
        boff += 128;
        ABU bu;
        bu.u[0] = b0.x; bu.u[1] = b0.y; bu.u[2] = b0.z; bu.u[3] = b0.w;
        bu.u[4] = b1.x; bu.u[5] = b1.y; bu.u[6] = b1.z; bu.u[7] = b1.w;

        acc = __builtin_amdgcn_wmma_f32_16x16x32_f16(
            false, au.v, false, bu.v, (short)0, acc, false, false);
        __syncthreads();
    }

    // combine K-halves, normalize, ELU, store
#pragma unroll
    for (int v = 0; v < 8; ++v) s_acc[wv][lane][v] = acc[v];
    __syncthreads();
    if (wv < 4) {
#pragma unroll
        for (int v = 0; v < 8; ++v) {
            const int Mr = v + hi * 8;
            float val = (s_acc[wv][lane][v] + s_acc[wv + 4][lane][v]) * s_inv[Mr];
            val = val > 0.f ? val : (__expf(val) - 1.0f);   // ELU(alpha=1)
            out[(size_t)(i0 + Mr) * FOUT + nt * 16 + (lane & 15)] = val;
        }
    }
}

extern "C" void kernel_launch(void* const* d_in, const int* in_sizes, int n_in,
                              void* d_out, int out_size, void* d_ws, size_t ws_size,
                              hipStream_t stream) {
    const float* h   = (const float*)d_in[0];
    const int*   adj = (const int*)d_in[1];
    const float* W   = (const float*)d_in[2];
    const float* a   = (const float*)d_in[3];
    float* out = (float*)d_out;

    char* ws = (char*)d_ws;
    float*    Wh = (float*)ws;                                   // 2 MB
    float*    fs = (float*)(ws + (size_t)2 * 1024 * 1024);       // 32 KB
    float*    fd = fs + NN;                                      // 32 KB
    unsigned* Wb = (unsigned*)(ws + (size_t)2 * 1024 * 1024 + 64 * 1024); // 1 MB

    k_wh<<<NN, 64, 0, stream>>>(h, W, a, Wh, fs, fd);
    k_swizzle_b<<<(4 * 256 * 32 + 255) / 256, 256, 0, stream>>>(Wh, Wb);
    k_attn<<<NN / 16, 256, 0, stream>>>(adj, Wb, fs, fd, out);
}
